// QuantumSuperpositionBranches_83863531421724
// MI455X (gfx1250) — matrix operand order repaired and checked by reading, not compiled, not run on hardware
//
#include <hip/hip_runtime.h>
#include <hip/hip_bf16.h>
#include <math.h>

#define NQ      12
#define NSTATE  4096          // 2^12
#define BATCH   128
#define KDIM    512
#define NPAR    240           // params per branch
#define NCOL    736           // 3*240 + 12 (x_q) + 4 pad
#define PI_F    3.14159265358979323846f

typedef __attribute__((ext_vector_type(16))) _Float16 v16h;
typedef __attribute__((ext_vector_type(8)))  _Float16 v8h;
typedef __attribute__((ext_vector_type(8)))  float    v8f;

// ---------------------------------------------------------------------------
// Prep: fp32 -> fp16 conversion of x; pack W1|W2|W3|Win into a single
// column-major (N x K) fp16 matrix so WMMA B-fragments are contiguous loads.
// ---------------------------------------------------------------------------
__global__ __launch_bounds__(256)
void cvt_x_kernel(const float* __restrict__ x, _Float16* __restrict__ xh, int n) {
  int i = blockIdx.x * blockDim.x + threadIdx.x;
  if (i < n) xh[i] = (_Float16)x[i];
}

__global__ __launch_bounds__(256)
void pack_w_kernel(const float* __restrict__ W1, const float* __restrict__ W2,
                   const float* __restrict__ W3, const float* __restrict__ Win,
                   _Float16* __restrict__ Bt) {
  // Bt[n*KDIM + k] = Wcat[k][n]   (column-major over n)
  int i = blockIdx.x * blockDim.x + threadIdx.x;
  if (i >= NCOL * KDIM) return;
  int n = i / KDIM;
  int k = i - n * KDIM;
  float v = 0.0f;
  if      (n < 240) v = W1[k * NPAR + n];
  else if (n < 480) v = W2[k * NPAR + (n - 240)];
  else if (n < 720) v = W3[k * NPAR + (n - 480)];
  else if (n < 732) v = Win[k * NQ + (n - 720)];
  Bt[i] = (_Float16)v;
}

// ---------------------------------------------------------------------------
// Fused GEMM: P[128 x 736] = x[128x512] @ Wcat[512x736] + bias
// One wave per 16x16 output tile, K swept in 32-element WMMA steps.
// Fragment layouts per CDNA5 ISA 7.12.2 (wave32).
// ---------------------------------------------------------------------------
__global__ __launch_bounds__(128)
void gemm_wmma_kernel(const _Float16* __restrict__ Ah,   // 128 x 512 row-major
                      const _Float16* __restrict__ Bt,   // NCOL x 512 (col-major over n)
                      const float* __restrict__ b1, const float* __restrict__ b2,
                      const float* __restrict__ b3, const float* __restrict__ b_in,
                      float* __restrict__ P) {
  const int lane  = threadIdx.x & 31;
  const int wave  = threadIdx.x >> 5;
  const int ntile = blockIdx.x * 4 + wave;   // 46 live N tiles
  const int mtile = blockIdx.y;              // 8 M tiles
  if (ntile >= (NCOL / 16)) return;          // wave-uniform guard (EXEC stays all-1s)

  const int row = lane & 15;
  const int kg  = lane >> 4;

  const _Float16* arow = Ah + (mtile * 16 + row) * KDIM + kg * 8;
  const _Float16* bcol = Bt + (ntile * 16 + row) * KDIM + kg * 8;

  v8f acc = {};
  for (int kk = 0; kk < KDIM; kk += 32) {
    __builtin_prefetch(arow + kk + 64, 0, 1);
    __builtin_prefetch(bcol + kk + 64, 0, 1);
    v8h alo = *(const v8h*)(arow + kk);
    v8h ahi = *(const v8h*)(arow + kk + 16);
    v8h blo = *(const v8h*)(bcol + kk);
    v8h bhi = *(const v8h*)(bcol + kk + 16);
    v16h a, b;
#pragma unroll
    for (int e = 0; e < 8; ++e) { a[e] = alo[e]; a[e + 8] = ahi[e]; }
#pragma unroll
    for (int e = 0; e < 8; ++e) { b[e] = blo[e]; b[e + 8] = bhi[e]; }
    acc = __builtin_amdgcn_wmma_f32_16x16x32_f16(
        /*neg_a=*/false, a, /*neg_b=*/false, b,
        /*c_mod=*/(short)0, acc, /*reuse_a=*/false, /*reuse_b=*/false);
  }

  const int ncol = ntile * 16 + row;         // N = lane&15 (== row)
  float bias = 0.0f;
  if      (ncol < 240) bias = b1[ncol];
  else if (ncol < 480) bias = b2[ncol - 240];
  else if (ncol < 720) bias = b3[ncol - 480];
  else if (ncol < 732) bias = b_in[ncol - 720];
#pragma unroll
  for (int r = 0; r < 8; ++r) {
    int m = mtile * 16 + kg * 8 + r;
    P[m * NCOL + ncol] = acc[r] + bias;
  }
}

// ---------------------------------------------------------------------------
// Complex 2x2 helpers for gate fusion (RZ*RY*RX[*RYenc] -> one unitary)
// ---------------------------------------------------------------------------
struct cmat2 { float2 m00, m01, m10, m11; };

__device__ __forceinline__ float2 cmul(float2 a, float2 b) {
  return make_float2(a.x * b.x - a.y * b.y, a.x * b.y + a.y * b.x);
}
__device__ __forceinline__ float2 cadd(float2 a, float2 b) {
  return make_float2(a.x + b.x, a.y + b.y);
}
__device__ __forceinline__ cmat2 mmul(const cmat2& A, const cmat2& B) {
  cmat2 C;
  C.m00 = cadd(cmul(A.m00, B.m00), cmul(A.m01, B.m10));
  C.m01 = cadd(cmul(A.m00, B.m01), cmul(A.m01, B.m11));
  C.m10 = cadd(cmul(A.m10, B.m00), cmul(A.m11, B.m10));
  C.m11 = cadd(cmul(A.m10, B.m01), cmul(A.m11, B.m11));
  return C;
}
__device__ __forceinline__ cmat2 rx_mat(float th) {
  float c = cosf(0.5f * th), s = sinf(0.5f * th);
  cmat2 M;
  M.m00 = make_float2(c, 0.f);  M.m01 = make_float2(0.f, -s);
  M.m10 = make_float2(0.f, -s); M.m11 = make_float2(c, 0.f);
  return M;
}
__device__ __forceinline__ cmat2 ry_mat(float th) {
  float c = cosf(0.5f * th), s = sinf(0.5f * th);
  cmat2 M;
  M.m00 = make_float2(c, 0.f); M.m01 = make_float2(-s, 0.f);
  M.m10 = make_float2(s, 0.f); M.m11 = make_float2(c, 0.f);
  return M;
}
__device__ __forceinline__ cmat2 rz_mat(float th) {
  float c = cosf(0.5f * th), s = sinf(0.5f * th);
  cmat2 M;
  M.m00 = make_float2(c, -s);   M.m01 = make_float2(0.f, 0.f);
  M.m10 = make_float2(0.f, 0.f); M.m11 = make_float2(c, s);
  return M;
}

__device__ __forceinline__ int pair_base(int p, int bit) {
  return ((p >> bit) << (bit + 1)) | (p & ((1 << bit) - 1));
}

// apply arbitrary single-qubit unitary to `wire`
__device__ __forceinline__ void apply_u(float2* st, int tid, int wire, const cmat2& U) {
  int bit = 11 - wire, stride = 1 << bit;
  for (int p = tid; p < NSTATE / 2; p += 256) {
    int i0 = pair_base(p, bit), i1 = i0 | stride;
    float2 a = st[i0], b = st[i1];
    float2 n0 = cadd(cmul(U.m00, a), cmul(U.m01, b));
    float2 n1 = cadd(cmul(U.m10, a), cmul(U.m11, b));
    st[i0] = n0; st[i1] = n1;
  }
  __syncthreads();
}

// controlled-RX (ctrl bit == 1)
__device__ __forceinline__ void apply_crx(float2* st, int tid, int ctrl, int tgt, float th) {
  float hc = cosf(0.5f * th), hs = sinf(0.5f * th);
  int bc = 11 - ctrl, bt = 11 - tgt;
  int lo = bc < bt ? bc : bt, hi = bc < bt ? bt : bc;
  for (int p = tid; p < NSTATE / 4; p += 256) {
    int t = pair_base(p, lo);
    t = pair_base(t, hi);
    int i0 = t | (1 << bc);           // control = 1, target = 0
    int i1 = i0 | (1 << bt);
    float2 a = st[i0], b = st[i1];
    float2 n0 = { hc * a.x + hs * b.y,  hc * a.y - hs * b.x };
    float2 n1 = { hs * a.y + hc * b.x, -hs * a.x + hc * b.y };
    st[i0] = n0; st[i1] = n1;
  }
  __syncthreads();
}

// ---------------------------------------------------------------------------
// 12-qubit statevector simulator: one block per (batch, branch).
// State (4096 complex64 = 32KB) lives in LDS for the whole circuit.
// ---------------------------------------------------------------------------
__global__ __launch_bounds__(256)
void quantum_kernel(const float* __restrict__ P,
                    const float* __restrict__ ar, const float* __restrict__ ai,
                    const float* __restrict__ br, const float* __restrict__ bi,
                    const float* __restrict__ gr, const float* __restrict__ gi,
                    float* __restrict__ out) {
  __shared__ float2 st[NSTATE];
  __shared__ float  prm[NPAR];
  __shared__ float  ang[NQ];
  __shared__ float  redx[8], redy[8], redz[8];

  const int tid    = threadIdx.x;
  const int batch  = blockIdx.x;
  const int branch = blockIdx.y;

  for (int i = tid; i < NSTATE; i += 256) st[i] = make_float2(0.0f, 0.0f);
  if (tid < NPAR) prm[tid] = P[batch * NCOL + branch * NPAR + tid];
  if (tid < NQ)   ang[tid] = tanhf(P[batch * NCOL + 720 + tid]) * PI_F;
  __syncthreads();
  if (tid == 0) st[0] = make_float2(1.0f, 0.0f);
  __syncthreads();

  // variational layers; encoding RY fused into layer-0 single-qubit unitaries
  int p = 0;
  for (int layer = 0; layer < 4; ++layer) {
    for (int i = 0; i < NQ; ++i) {
      // U = RZ(p+2) * RY(p+1) * RX(p) [* RYenc(ang_i) for layer 0]
      cmat2 U = mmul(rz_mat(prm[p + 2]), mmul(ry_mat(prm[p + 1]), rx_mat(prm[p])));
      if (layer == 0) U = mmul(U, ry_mat(ang[i]));
      apply_u(st, tid, i, U);
      p += 3;
    }
    for (int i = 0; i < NQ; ++i)      { apply_crx(st, tid, i, (i + 1) % NQ, prm[p]); ++p; }
    for (int i = NQ - 1; i >= 0; --i) { apply_crx(st, tid, i, (i + 11) % NQ, prm[p]); ++p; }
  }

  // branch-mixing scale: |w_branch| / sqrt(|a|^2+|b|^2+|g|^2+1e-8)
  const float na = ar[0] * ar[0] + ai[0] * ai[0];
  const float nb = br[0] * br[0] + bi[0] * bi[0];
  const float ng = gr[0] * gr[0] + gi[0] * gi[0];
  const float norm = sqrtf(na + nb + ng + 1e-8f);
  const float wsel = (branch == 0) ? sqrtf(na) : (branch == 1) ? sqrtf(nb) : sqrtf(ng);
  const float scale = wsel / norm;

  // Pauli X/Y/Z expectations per qubit
  for (int q = 0; q < NQ; ++q) {
    const int bit = 11 - q, stride = 1 << bit;
    float lx = 0.0f, ly = 0.0f, lz = 0.0f;
    for (int pp = tid; pp < NSTATE / 2; pp += 256) {
      int i0 = pair_base(pp, bit), i1 = i0 | stride;
      float2 a = st[i0], b = st[i1];
      float cr = a.x * b.x + a.y * b.y;     // Re(conj(a0)*a1)
      float ci = a.x * b.y - a.y * b.x;     // Im(conj(a0)*a1)
      lx += 2.0f * cr;
      ly += 2.0f * ci;
      lz += (a.x * a.x + a.y * a.y) - (b.x * b.x + b.y * b.y);
    }
#pragma unroll
    for (int off = 16; off > 0; off >>= 1) {
      lx += __shfl_down(lx, off);
      ly += __shfl_down(ly, off);
      lz += __shfl_down(lz, off);
    }
    if ((tid & 31) == 0) { redx[tid >> 5] = lx; redy[tid >> 5] = ly; redz[tid >> 5] = lz; }
    __syncthreads();
    if (tid == 0) {
      float sx = 0.0f, sy = 0.0f, sz = 0.0f;
      for (int w = 0; w < 8; ++w) { sx += redx[w]; sy += redy[w]; sz += redz[w]; }
      float* orow = out + batch * (9 * NQ) + branch * (3 * NQ);
      orow[q]          = scale * sx;
      orow[NQ + q]     = scale * sy;
      orow[2 * NQ + q] = scale * sz;
    }
    __syncthreads();
  }
}

// ---------------------------------------------------------------------------
extern "C" void kernel_launch(void* const* d_in, const int* in_sizes, int n_in,
                              void* d_out, int out_size, void* d_ws, size_t ws_size,
                              hipStream_t stream) {
  const float* x    = (const float*)d_in[0];
  const float* W1   = (const float*)d_in[1];
  const float* b1   = (const float*)d_in[2];
  const float* W2   = (const float*)d_in[3];
  const float* b2   = (const float*)d_in[4];
  const float* W3   = (const float*)d_in[5];
  const float* b3   = (const float*)d_in[6];
  const float* Win  = (const float*)d_in[7];
  const float* b_in = (const float*)d_in[8];
  const float* a_r  = (const float*)d_in[9];
  const float* a_i  = (const float*)d_in[10];
  const float* be_r = (const float*)d_in[11];
  const float* be_i = (const float*)d_in[12];
  const float* g_r  = (const float*)d_in[13];
  const float* g_i  = (const float*)d_in[14];
  float* out = (float*)d_out;

  // workspace layout
  char* ws = (char*)d_ws;
  _Float16* xh = (_Float16*)ws;                               // 128*512*2  = 131072 B
  _Float16* Bt = (_Float16*)(ws + 131072);                    // 736*512*2  = 753664 B
  float*    P  = (float*)(ws + 131072 + 753664);              // 128*736*4  = 376832 B

  // 1) convert x to f16
  cvt_x_kernel<<<(BATCH * KDIM + 255) / 256, 256, 0, stream>>>(x, xh, BATCH * KDIM);
  // 2) pack + transpose weights to f16, col-major
  pack_w_kernel<<<(NCOL * KDIM + 255) / 256, 256, 0, stream>>>(W1, W2, W3, Win, Bt);
  // 3) fused WMMA GEMM (+bias): P = x @ [W1|W2|W3|Win]
  gemm_wmma_kernel<<<dim3((NCOL / 16 + 3) / 4, BATCH / 16), 128, 0, stream>>>(
      xh, Bt, b1, b2, b3, b_in, P);
  // 4) statevector simulation, 3 branches x 128 batch, output scaled in place
  quantum_kernel<<<dim3(BATCH, 3), 256, 0, stream>>>(
      P, a_r, a_i, be_r, be_i, g_r, g_i, out);
}
